// MambaEncoder_12146167513464
// MI455X (gfx1250) — matrix operand order repaired and checked by reading, not compiled, not run on hardware
//
#include <hip/hip_runtime.h>
#include <hip/hip_bf16.h>

// ---------------- problem constants ----------------
#define B_      2
#define L_      2048
#define DM      512          // D_MODEL
#define DI      1024         // D_INNER
#define DS      16           // D_STATE
#define DTR     32           // DT_RANK
#define NB      4
#define DBC_N   (DTR + 2*DS) // 64
#define MROWS   (B_ * L_)    // 4096
#define CH      16           // scan chunks along L
#define CLEN    (L_ / CH)    // 128

// ---------------- vector types ----------------
typedef __attribute__((ext_vector_type(16))) __bf16 v16bf;
typedef __attribute__((ext_vector_type(8)))  __bf16 v8bf;
typedef __attribute__((ext_vector_type(4)))  __bf16 v4bf;
typedef __attribute__((ext_vector_type(8)))  float  v8f;

static __device__ __forceinline__ __bf16 f2bf(float f) {
  unsigned u = __builtin_bit_cast(unsigned, f);
  u += 0x7FFFu + ((u >> 16) & 1u);            // round-to-nearest-even
  unsigned short s = (unsigned short)(u >> 16);
  return __builtin_bit_cast(__bf16, s);
}

static __device__ __forceinline__ v16bf cat16(v8bf lo, v8bf hi) {
  return __builtin_shufflevector(lo, hi, 0,1,2,3,4,5,6,7,8,9,10,11,12,13,14,15);
}

// A fragment: 16x32 bf16 (row-major A[M,K]), row = lane%16.
static __device__ __forceinline__ v16bf load_a_frag(const __bf16* __restrict__ A,
                                                    int row, int lda, int k0, int half) {
  const __bf16* p = A + (size_t)row * lda + k0 + half * 8;
  v8bf lo = *(const v8bf*)p;          // 16B
  v8bf hi = *(const v8bf*)(p + 16);   // 16B
  return cat16(lo, hi);
}

// B fragment from pre-transposed Bt[N,K] bf16, col = lane%16.
static __device__ __forceinline__ v16bf load_b_frag(const __bf16* __restrict__ Bt,
                                                    int col, int ldb, int k0, int half) {
  const __bf16* p = Bt + (size_t)col * ldb + k0 + half * 8;
  v8bf lo = *(const v8bf*)p;
  v8bf hi = *(const v8bf*)(p + 16);
  return cat16(lo, hi);
}

// ---------------- bf16 WMMA GEMM, fp32 accumulate ----------------
// C[M,N] = A[M,K] * Bt[N,K]^T   (K % 32 == 0, M % 64 == 0)
// EPI: 0 = none, 1 = softplus(acc + bias[n]), 2 = acc + resid[m*ldc+n]
// WBF: also store bf16 copy of C to Ch
template <int EPI, bool WBF>
__global__ __launch_bounds__(256) void gemm_bf16_wmma(
    const __bf16* __restrict__ A, const __bf16* __restrict__ Bt,
    float* __restrict__ C, __bf16* __restrict__ Ch,
    int M, int N, int K, int lda, int ldb, int ldc,
    const float* __restrict__ bias, const float* __restrict__ resid) {
  const int tid  = threadIdx.x;
  const int lane = tid & 31;
  const int w    = tid >> 5;
  const int wy   = w >> 2;
  const int wx   = w & 3;
  const int half = lane >> 4;
  const int l16  = lane & 15;

  const int row0 = blockIdx.y * 64 + wy * 32;
  const int col0 = blockIdx.x * 128 + wx * 32;

  const int ra0 = min(row0 + l16, M - 1);
  const int ra1 = min(row0 + 16 + l16, M - 1);
  const int cb0 = min(col0 + l16, N - 1);
  const int cb1 = min(col0 + 16 + l16, N - 1);

  v8f acc00 = {}, acc01 = {}, acc10 = {}, acc11 = {};

  for (int k0 = 0; k0 < K; k0 += 32) {
    v16bf a0 = load_a_frag(A, ra0, lda, k0, half);
    v16bf a1 = load_a_frag(A, ra1, lda, k0, half);
    v16bf b0 = load_b_frag(Bt, cb0, ldb, k0, half);
    v16bf b1 = load_b_frag(Bt, cb1, ldb, k0, half);
    acc00 = __builtin_amdgcn_wmma_f32_16x16x32_bf16(false, a0, false, b0, (short)0, acc00, false, false);
    acc01 = __builtin_amdgcn_wmma_f32_16x16x32_bf16(false, a0, false, b1, (short)0, acc01, false, false);
    acc10 = __builtin_amdgcn_wmma_f32_16x16x32_bf16(false, a1, false, b0, (short)0, acc10, false, false);
    acc11 = __builtin_amdgcn_wmma_f32_16x16x32_bf16(false, a1, false, b1, (short)0, acc11, false, false);
  }

#pragma unroll
  for (int i = 0; i < 2; ++i) {
#pragma unroll
    for (int j = 0; j < 2; ++j) {
      const v8f acc = (i == 0) ? (j == 0 ? acc00 : acc01) : (j == 0 ? acc10 : acc11);
      const int n = col0 + j * 16 + l16;
      if (n >= N) continue;
#pragma unroll
      for (int r = 0; r < 8; ++r) {
        const int m = row0 + i * 16 + half * 8 + r;
        if (m >= M) continue;
        float v = acc[r];
        if (EPI == 1) {
          v += bias[n];
          v = (v > 20.f) ? v : __logf(1.f + __expf(v));   // softplus
        } else if (EPI == 2) {
          v += resid[(size_t)m * ldc + n];
        }
        C[(size_t)m * ldc + n] = v;
        if (WBF) Ch[(size_t)m * ldc + n] = f2bf(v);
      }
    }
  }
}

// ---------------- weight convert + transpose: in[R,C] f32 -> out[C,R] bf16 ----
__global__ __launch_bounds__(256) void transpose_to_bf16(
    const float* __restrict__ in, __bf16* __restrict__ out, int R, int C) {
  __shared__ __bf16 tile[32][33];
  const int tx = threadIdx.x, ty = threadIdx.y;      // (32,8)
  const int c0 = blockIdx.x * 32, r0 = blockIdx.y * 32;
#pragma unroll
  for (int j = 0; j < 4; ++j)
    tile[ty + j * 8][tx] = f2bf(in[(size_t)(r0 + ty + j * 8) * C + c0 + tx]);
  __syncthreads();
#pragma unroll
  for (int j = 0; j < 4; ++j)
    out[(size_t)(c0 + ty + j * 8) * R + r0 + tx] = tile[tx][ty + j * 8];
}

// ---------------- flat f32 -> bf16 convert (x4 vectorized) ----------------
__global__ __launch_bounds__(256) void convert_bf16(
    const float* __restrict__ in, __bf16* __restrict__ out, int n4) {
  const int i = blockIdx.x * 256 + threadIdx.x;
  if (i >= n4) return;
  const float4 v = ((const float4*)in)[i];
  v4bf o; o[0] = f2bf(v.x); o[1] = f2bf(v.y); o[2] = f2bf(v.z); o[3] = f2bf(v.w);
  ((v4bf*)out)[i] = o;
}

// ---------------- depthwise causal conv (width 4) + bias + SiLU ----------------
__global__ __launch_bounds__(256) void conv_silu_kernel(
    const float* __restrict__ xz, const float* __restrict__ cw,
    const float* __restrict__ cb, float* __restrict__ xc_f,
    __bf16* __restrict__ xc_h) {
  const int idx = blockIdx.x * 256 + threadIdx.x;       // over MROWS*DI
  const int d  = idx & (DI - 1);
  const int bl = idx >> 10;
  const int l  = bl & (L_ - 1);

  const float w0 = cw[d * 4 + 0], w1 = cw[d * 4 + 1];
  const float w2 = cw[d * 4 + 2], w3 = cw[d * 4 + 3];
  const float* p = xz + (size_t)bl * (2 * DI) + d;

  float acc = cb[d] + p[0] * w3;
  if (l >= 1) acc += p[-1 * 2 * DI] * w2;
  if (l >= 2) acc += p[-2 * 2 * DI] * w1;
  if (l >= 3) acc += p[-3 * 2 * DI] * w0;

  const float s = acc / (1.f + __expf(-acc));           // SiLU
  xc_f[idx] = s;
  xc_h[idx] = f2bf(s);
}

// ================== chunked selective scan (3 passes) ==================
// Recurrence h[l] = dA[l]*h[l-1] + dBx[l], split into CH chunks of CLEN.
// Thread layout per block: 256 threads = 16 channels x 16 states.
// blockIdx.x (pass1/3) = (b*64 + dblk)*CH + c  -> 2048 blocks, 16K waves.

// Pass 1: per-chunk transfer function (A_ch = prod dA, B_ch = local end state)
__global__ __launch_bounds__(256) void scan_pass1(
    const float* __restrict__ dt, const float* __restrict__ xc,
    const float* __restrict__ dbc, const float* __restrict__ A_log,
    float* __restrict__ Ach, float* __restrict__ Bch) {
  const int c    = blockIdx.x & (CH - 1);
  const int bd   = blockIdx.x / CH;                    // b*64 + dblk
  const int b    = bd >> 6;
  const int dblk = bd & 63;
  const int tid  = threadIdx.x;
  const int s    = tid & 15;
  const int d    = dblk * 16 + (tid >> 4);

  const float Ads = -__expf(A_log[d * DS + s]);
  float h = 0.f, aprod = 1.f;
  const size_t blBase = (size_t)b * L_ + (size_t)c * CLEN;

  for (int l = 0; l < CLEN; ++l) {
    const size_t bl = blBase + l;
    const float dtv = dt[bl * DI + d];
    const float xcv = xc[bl * DI + d];
    const float Bv  = dbc[bl * DBC_N + DTR + s];
    const float dA  = __expf(dtv * Ads);
    h = dA * h + (dtv * xcv) * Bv;
    aprod *= dA;
  }
  const size_t idx = (size_t)blockIdx.x * 256 + tid;
  Ach[idx] = aprod;
  Bch[idx] = h;
}

// Pass 2: sequentially combine CH chunk transfers -> incoming state per chunk
__global__ __launch_bounds__(256) void scan_pass2(
    const float* __restrict__ Ach, const float* __restrict__ Bch,
    float* __restrict__ Hinit) {
  const int bd  = blockIdx.x;                          // b*64 + dblk
  const int tid = threadIdx.x;
  float h = 0.f;
#pragma unroll
  for (int c = 0; c < CH; ++c) {
    const size_t idx = ((size_t)bd * CH + c) * 256 + tid;
    Hinit[idx] = h;
    h = Ach[idx] * h + Bch[idx];
  }
}

// Pass 3: replay chunk from Hinit, reduce 16 states with shuffles, fuse
// y = (sum_s h*C) + xc*Dp, gated by silu(z); store bf16.
__global__ __launch_bounds__(256) void scan_pass3(
    const float* __restrict__ dt, const float* __restrict__ xc,
    const float* __restrict__ dbc, const float* __restrict__ xz,
    const float* __restrict__ A_log, const float* __restrict__ Dp,
    const float* __restrict__ Hinit, __bf16* __restrict__ y) {
  const int c    = blockIdx.x & (CH - 1);
  const int bd   = blockIdx.x / CH;
  const int b    = bd >> 6;
  const int dblk = bd & 63;
  const int tid  = threadIdx.x;
  const int s    = tid & 15;
  const int d    = dblk * 16 + (tid >> 4);

  const float Ads = -__expf(A_log[d * DS + s]);
  const float Dd  = Dp[d];
  float h = Hinit[(size_t)blockIdx.x * 256 + tid];
  const size_t blBase = (size_t)b * L_ + (size_t)c * CLEN;

  for (int l = 0; l < CLEN; ++l) {
    const size_t bl = blBase + l;
    if (l + 8 < CLEN) {                                // stream-ahead hints
      __builtin_prefetch(&dt[(bl + 8) * DI + d], 0, 1);
      __builtin_prefetch(&xc[(bl + 8) * DI + d], 0, 1);
    }
    const float dtv = dt[bl * DI + d];
    const float xcv = xc[bl * DI + d];
    const float Bv  = dbc[bl * DBC_N + DTR + s];
    const float Cv  = dbc[bl * DBC_N + DTR + DS + s];

    const float dA = __expf(dtv * Ads);
    h = dA * h + (dtv * xcv) * Bv;

    float p = h * Cv;                                  // reduce over 16 states
    p += __shfl_xor(p, 1);
    p += __shfl_xor(p, 2);
    p += __shfl_xor(p, 4);
    p += __shfl_xor(p, 8);

    if (s == 0) {
      const float zv = xz[bl * (2 * DI) + DI + d];
      float yv = p + xcv * Dd;
      yv *= zv / (1.f + __expf(-zv));                  // * silu(z)
      y[bl * DI + d] = f2bf(yv);
    }
  }
}

// ---------------- host-side launch ----------------
extern "C" void kernel_launch(void* const* d_in, const int* in_sizes, int n_in,
                              void* d_out, int out_size, void* d_ws, size_t ws_size,
                              hipStream_t stream) {
  const float* x0     = (const float*)d_in[0];
  const float* W_in   = (const float*)d_in[1];
  const float* conv_w = (const float*)d_in[2];
  const float* conv_b = (const float*)d_in[3];
  const float* W_x    = (const float*)d_in[4];
  const float* W_dt   = (const float*)d_in[5];
  const float* b_dt   = (const float*)d_in[6];
  const float* A_log  = (const float*)d_in[7];
  const float* Dp     = (const float*)d_in[8];
  const float* W_out  = (const float*)d_in[9];

  char* ws = (char*)d_ws;
  size_t off = 0;
  auto carve = [&](size_t bytes) { void* p = ws + off; off += (bytes + 255) & ~(size_t)255; return p; };

  float*  xz_f  = (float*) carve((size_t)MROWS * 2 * DI * 4);
  float*  xc_f  = (float*) carve((size_t)MROWS * DI * 4);
  float*  dt_f  = (float*) carve((size_t)MROWS * DI * 4);
  float*  dbc_f = (float*) carve((size_t)MROWS * DBC_N * 4);
  float*  xA_f  = (float*) carve((size_t)MROWS * DM * 4);
  float*  xB_f  = (float*) carve((size_t)MROWS * DM * 4);
  __bf16* yb_h  = (__bf16*)carve((size_t)MROWS * DI * 2);
  __bf16* xc_h  = (__bf16*)carve((size_t)MROWS * DI * 2);
  __bf16* dbc_h = (__bf16*)carve((size_t)MROWS * DBC_N * 2);
  __bf16* xA_h  = (__bf16*)carve((size_t)MROWS * DM * 2);
  __bf16* xB_h  = (__bf16*)carve((size_t)MROWS * DM * 2);
  __bf16* x0_h  = (__bf16*)carve((size_t)MROWS * DM * 2);
  __bf16* WinT  = (__bf16*)carve((size_t)NB * DM * 2 * DI * 2);
  __bf16* WxT   = (__bf16*)carve((size_t)NB * DI * DBC_N * 2);
  __bf16* WdtT  = (__bf16*)carve((size_t)NB * DTR * DI * 2);
  __bf16* WoutT = (__bf16*)carve((size_t)NB * DI * DM * 2);
  float*  Ach   = (float*) carve((size_t)B_ * 64 * CH * 256 * 4);  // chunk transfers
  float*  Bch   = (float*) carve((size_t)B_ * 64 * CH * 256 * 4);
  float*  Hini  = (float*) carve((size_t)B_ * 64 * CH * 256 * 4);

  dim3 t32x8(32, 8);
  convert_bf16<<<(MROWS * DM / 4 + 255) / 256, 256, 0, stream>>>(x0, x0_h, MROWS * DM / 4);
  for (int i = 0; i < NB; ++i) {
    transpose_to_bf16<<<dim3(2 * DI / 32, DM / 32), t32x8, 0, stream>>>(
        W_in + (size_t)i * DM * 2 * DI, WinT + (size_t)i * DM * 2 * DI, DM, 2 * DI);
    transpose_to_bf16<<<dim3(DBC_N / 32, DI / 32), t32x8, 0, stream>>>(
        W_x + (size_t)i * DI * DBC_N, WxT + (size_t)i * DI * DBC_N, DI, DBC_N);
    transpose_to_bf16<<<dim3(DI / 32, DTR / 32), t32x8, 0, stream>>>(
        W_dt + (size_t)i * DTR * DI, WdtT + (size_t)i * DTR * DI, DTR, DI);
    transpose_to_bf16<<<dim3(DM / 32, DI / 32), t32x8, 0, stream>>>(
        W_out + (size_t)i * DI * DM, WoutT + (size_t)i * DI * DM, DI, DM);
  }

  const int M = MROWS;
  for (int i = 0; i < NB; ++i) {
    const __bf16* Acur_h  = (i == 0) ? x0_h : ((i & 1) ? xA_h : xB_h);
    const float*  resid_f = (i == 0) ? x0   : ((i & 1) ? xA_f : xB_f);
    float*        xnext_f = (i == NB - 1) ? (float*)d_out : ((i & 1) ? xB_f : xA_f);
    __bf16*       xnext_h = (i & 1) ? xB_h : xA_h;

    const __bf16* WinT_i  = WinT  + (size_t)i * DM * 2 * DI;
    const __bf16* WxT_i   = WxT   + (size_t)i * DI * DBC_N;
    const __bf16* WdtT_i  = WdtT  + (size_t)i * DTR * DI;
    const __bf16* WoutT_i = WoutT + (size_t)i * DI * DM;
    const float*  cwi = conv_w + (size_t)i * DI * 4;
    const float*  cbi = conv_b + (size_t)i * DI;
    const float*  bdi = b_dt   + (size_t)i * DI;
    const float*  Ali = A_log  + (size_t)i * DI * DS;
    const float*  Dpi = Dp     + (size_t)i * DI;

    // 1) xz = x @ W_in   [M,512] x [512,2048]
    gemm_bf16_wmma<0, false><<<dim3((2 * DI) / 128, M / 64), 256, 0, stream>>>(
        Acur_h, WinT_i, xz_f, nullptr, M, 2 * DI, DM, DM, DM, 2 * DI, nullptr, nullptr);

    // 2) depthwise conv + SiLU -> xc (f32 + bf16)
    conv_silu_kernel<<<(M * DI) / 256, 256, 0, stream>>>(xz_f, cwi, cbi, xc_f, xc_h);

    // 3) dbc = xc @ W_x   [M,1024] x [1024,64]
    gemm_bf16_wmma<0, true><<<dim3(1, M / 64), 256, 0, stream>>>(
        xc_h, WxT_i, dbc_f, dbc_h, M, DBC_N, DI, DI, DI, DBC_N, nullptr, nullptr);

    // 4) dt = softplus(dt_raw @ W_dt + b_dt)   [M,32] x [32,1024]
    gemm_bf16_wmma<1, false><<<dim3(DI / 128, M / 64), 256, 0, stream>>>(
        dbc_h, WdtT_i, dt_f, nullptr, M, DI, DTR, DBC_N, DTR, DI, bdi, nullptr);

    // 5) chunked selective scan -> y (bf16)
    scan_pass1<<<B_ * 64 * CH, 256, 0, stream>>>(dt_f, xc_f, dbc_f, Ali, Ach, Bch);
    scan_pass2<<<B_ * 64, 256, 0, stream>>>(Ach, Bch, Hini);
    scan_pass3<<<B_ * 64 * CH, 256, 0, stream>>>(dt_f, xc_f, dbc_f, xz_f, Ali, Dpi, Hini, yb_h);

    // 6) x = y @ W_out + residual   [M,1024] x [1024,512]
    if (i == NB - 1) {
      gemm_bf16_wmma<2, false><<<dim3(DM / 128, M / 64), 256, 0, stream>>>(
          yb_h, WoutT_i, xnext_f, nullptr, M, DM, DI, DI, DI, DM, nullptr, resid_f);
    } else {
      gemm_bf16_wmma<2, true><<<dim3(DM / 128, M / 64), 256, 0, stream>>>(
          yb_h, WoutT_i, xnext_f, xnext_h, M, DM, DI, DI, DI, DM, nullptr, resid_f);
    }
  }
}
